// EntitiyReasoningNetwork_84026740179017
// MI455X (gfx1250) — matrix operand order repaired and checked by reading, not compiled, not run on hardware
//
#include <hip/hip_runtime.h>

// Sizes from the reference
#define HS 1024   // lstm hidden
#define VS 512    // action vocab
#define WVS 300   // word vec
#define SS 64     // steps
#define NS 512    // entities

#define SCAN_BLOCKS 128   // persistent-kernel grid (trivially co-resident)

typedef __attribute__((ext_vector_type(16))) __bf16 v16bf;
typedef __attribute__((ext_vector_type(8)))  float  v8f;

__device__ __forceinline__ unsigned short f2bf_rne(float x) {
    unsigned int u = __float_as_uint(x);
    unsigned int r = (u + 0x7FFFu + ((u >> 16) & 1u)) >> 16;
    return (unsigned short)r;
}

__device__ __forceinline__ float wave_sum(float v) {
#pragma unroll
    for (int off = 16; off > 0; off >>= 1) v += __shfl_xor(v, off, 32);
    return v;
}

// Device-scope grid barrier (fixed, co-resident grid): L2 atomic arrive +
// generation-count spin with s_sleep. Acquire/release at agent scope give the
// cross-WGP visibility (L0 invalidate / L2 ordering) between scan phases.
__device__ __forceinline__ void grid_barrier(unsigned int* cnt, unsigned int* gen,
                                             unsigned int nblocks) {
    __syncthreads();
    if (threadIdx.x == 0) {
        unsigned int g = __hip_atomic_load(gen, __ATOMIC_RELAXED, __HIP_MEMORY_SCOPE_AGENT);
        unsigned int a = __hip_atomic_fetch_add(cnt, 1u, __ATOMIC_ACQ_REL, __HIP_MEMORY_SCOPE_AGENT);
        if (a == nblocks - 1u) {
            __hip_atomic_store(cnt, 0u, __ATOMIC_RELAXED, __HIP_MEMORY_SCOPE_AGENT);
            __hip_atomic_store(gen, g + 1u, __ATOMIC_RELEASE, __HIP_MEMORY_SCOPE_AGENT);
        } else {
            while (__hip_atomic_load(gen, __ATOMIC_ACQUIRE, __HIP_MEMORY_SCOPE_AGENT) == g) {
                __builtin_amdgcn_s_sleep(2);
            }
        }
    }
    __syncthreads();
}

// ---------------------------------------------------------------------------
// fp32 -> bf16 elementwise convert
// ---------------------------------------------------------------------------
__global__ void __launch_bounds__(256)
k_f2bf(const float* __restrict__ in, unsigned short* __restrict__ out, int n) {
    int i = blockIdx.x * 256 + threadIdx.x;
    if (i < n) out[i] = f2bf_rne(in[i]);
}

__global__ void k_init_sync(unsigned int* p) {
    if (threadIdx.x < 2) p[threadIdx.x] = 0u;
}

// ---------------------------------------------------------------------------
// Generic WMMA GEMM:  out[m, j] = act( sum_k A[m,k] * B[j,k] + bias[j] )
//   A: [64, K] bf16 row-major, B: [Nout, K] bf16 row-major (weights [out,in])
//   M fixed = 64 -> 4 m-tiles; one wave per 16x16 output tile.
//   act: 0 = none, 1 = relu, 2 = sigmoid
// ---------------------------------------------------------------------------
union BF16x16 { v16bf v; uint4 q[2]; };

__global__ void __launch_bounds__(256)
wmma_gemm_bf16(const unsigned short* __restrict__ A,
               const unsigned short* __restrict__ B,
               const float* __restrict__ bias,
               float* __restrict__ outf, int ldf,
               unsigned short* __restrict__ outbf, int ldbf, int coloff,
               int K, int ntiles, int act) {
    int tile = blockIdx.x * 8 + (threadIdx.x >> 5);
    if (tile >= 4 * ntiles) return;           // whole wave exits -> EXEC all-1s for WMMA
    int lane = threadIdx.x & 31;
    int hi   = lane >> 4;
    int l15  = lane & 15;
    int mt   = tile & 3;
    int nt   = tile >> 2;

    int arow = mt * 16 + l15;                 // A: lane holds one M row
    int jcol = nt * 16 + l15;                 // B/C/D: lane holds one N column
    const unsigned short* arp = A + (size_t)arow * K;
    const unsigned short* brp = B + (size_t)jcol * K;

    v8f c = {0.f, 0.f, 0.f, 0.f, 0.f, 0.f, 0.f, 0.f};
    for (int k0 = 0; k0 < K; k0 += 32) {
        BF16x16 a, b;
        // A 16x32 bf16: lanes 0-15 hold K {0..7,16..23}, lanes 16-31 {8..15,24..31}
        a.q[0] = *(const uint4*)(arp + k0 + (hi ? 8  : 0));
        a.q[1] = *(const uint4*)(arp + k0 + (hi ? 24 : 16));
        // B 32x16 bf16: lane = column, lanes 0-15 hold K {0..7,16..23}, lanes 16-31 {8..15,24..31}
        const unsigned short* bp = brp + k0 + (hi ? 8 : 0);
        b.q[0] = *(const uint4*)(bp);
        b.q[1] = *(const uint4*)(bp + 16);
        c = __builtin_amdgcn_wmma_f32_16x16x32_bf16(false, a.v, false, b.v,
                                                    (short)0, c, false, false);
    }

    float bs = bias[jcol];
#pragma unroll
    for (int r = 0; r < 8; ++r) {
        int   m = mt * 16 + r + hi * 8;       // C/D: VGPR r -> row r (lo) / r+8 (hi)
        float v = c[r] + bs;
        if (act == 1)      v = v > 0.f ? v : 0.f;
        else if (act == 2) v = 1.f / (1.f + __expf(-v));
        if (outf)  outf[(size_t)m * ldf + jcol] = v;
        if (outbf) outbf[(size_t)m * ldbf + coloff + jcol] = f2bf_rne(v);
    }
}

// ---------------------------------------------------------------------------
// choice = softmax(hat @ W3^T + W3b); keep c0, c1 (c2 multiplies zero)
// ---------------------------------------------------------------------------
__global__ void __launch_bounds__(32)
k_choice(const float* __restrict__ hat, const float* __restrict__ W3w,
         const float* __restrict__ W3b, float* __restrict__ c0,
         float* __restrict__ c1) {
    int s = blockIdx.x, lane = threadIdx.x;
    const float* hp = hat + s * HS;
    float a0 = 0.f, a1 = 0.f, a2 = 0.f;
    for (int k = lane; k < HS; k += 32) {
        float hv = hp[k];
        a0 += hv * W3w[k];
        a1 += hv * W3w[HS + k];
        a2 += hv * W3w[2 * HS + k];
    }
    a0 = wave_sum(a0); a1 = wave_sum(a1); a2 = wave_sum(a2);
    if (lane == 0) {
        a0 += W3b[0]; a1 += W3b[1]; a2 += W3b[2];
        float m  = fmaxf(a0, fmaxf(a1, a2));
        float e0 = __expf(a0 - m), e1 = __expf(a1 - m), e2 = __expf(a2 - m);
        float inv = 1.f / (e0 + e1 + e2);
        c0[s] = e0 * inv;
        c1[s] = e1 * inv;
    }
}

// row sums of ac_prob [S, V]
__global__ void __launch_bounds__(32)
k_acrow(const float* __restrict__ ac, float* __restrict__ acsum) {
    int s = blockIdx.x, lane = threadIdx.x;
    float a = 0.f;
    for (int k = lane; k < VS; k += 32) a += ac[s * VS + k];
    a = wave_sum(a);
    if (lane == 0) acsum[s] = a;
}

// bar_ft[s,w] = (sum_k ac[s,k] * E[k,w]) / acsum[s]
__global__ void __launch_bounds__(128)
k_barft(const float* __restrict__ ac, const float* __restrict__ E,
        const float* __restrict__ acsum, float* __restrict__ barft) {
    int s = blockIdx.y;
    int w = blockIdx.x * 128 + threadIdx.x;
    if (w >= WVS) return;
    const float* ap = ac + s * VS;
    float acc = 0.f;
    for (int k = 0; k < VS; ++k) acc += ap[k] * E[k * WVS + w];
    barft[s * WVS + w] = acc / acsum[s];
}

// scal[s] = bar_ft[s] . W4w + W4b
__global__ void __launch_bounds__(32)
k_scal(const float* __restrict__ barft, const float* __restrict__ W4w,
       const float* __restrict__ W4b, float* __restrict__ scal) {
    int s = blockIdx.x, lane = threadIdx.x;
    float a = 0.f;
    for (int w = lane; w < WVS; w += 32) a += barft[s * WVS + w] * W4w[w];
    a = wave_sum(a);
    if (lane == 0) scal[s] = a + W4b[0];
}

// ---------------------------------------------------------------------------
// Persistent scan kernel: all 64 steps in one launch.
// Three phases per step, separated by device-scope grid barriers:
//   P1: ent_prob[i] = sigmoid(ev[i].w2v_t); attn[i] = c0*p + c1*prev_p
//   P2: bar_et[h] = sum_i attn[i]*ev[i,h] / sum(attn); kt = relu(scal*bar_et)
//   P3: ev_new = attn (x) kt + (1-attn) (x) ev   -> step_all[t] (next step's ev)
// ev is 2 MB -> fully L2-resident; phases are pure L2-bandwidth streams.
// ---------------------------------------------------------------------------
__global__ void __launch_bounds__(256)
scan_persistent(const float* __restrict__ ent, const float* __restrict__ w2v,
                const float* __restrict__ c0, const float* __restrict__ c1,
                const float* __restrict__ scal,
                float* __restrict__ o_entp, float* __restrict__ o_sent,
                float* __restrict__ o_sall,
                float* __restrict__ attn, float* __restrict__ kt,
                unsigned int* __restrict__ sync) {
    const unsigned int nb = gridDim.x;
    unsigned int* cnt = sync;
    unsigned int* gen = sync + 1;
    const int lane = threadIdx.x & 31;
    const int wid  = blockIdx.x * 8 + (threadIdx.x >> 5);   // global wave id
    const int tid  = blockIdx.x * 256 + threadIdx.x;        // global thread id

    for (int t = 0; t < SS; ++t) {
        const float* ev = (t == 0) ? ent : (o_sall + (size_t)(t - 1) * NS * HS);

        // ---- P1: one wave per entity row (float4-vectorized dot) ----
        const float4* w4 = (const float4*)(w2v + t * HS);
        for (int i = wid; i < NS; i += nb * 8) {
            const float4* evp4 = (const float4*)(ev + (size_t)i * HS);
            float acc = 0.f;
            for (int k = lane; k < HS / 4; k += 32) {
                float4 e = evp4[k], w = w4[k];
                acc += e.x * w.x + e.y * w.y + e.z * w.z + e.w * w.w;
            }
            acc = wave_sum(acc);
            if (lane == 0) {
                float p  = 1.f / (1.f + __expf(-acc));
                float pv = (t == 0) ? 0.f : o_entp[(t - 1) * NS + i];
                o_entp[t * NS + i] = p;
                attn[i] = c0[t] * p + c1[t] * pv;
            }
        }
        grid_barrier(cnt, gen, nb);

        // ---- P2: one thread per hidden column ----
        for (int h = tid; h < HS; h += nb * 256) {
            float acc = 0.f, asum = 0.f;
            for (int i = 0; i < NS; ++i) {
                float a = attn[i];
                asum += a;
                acc  += a * ev[(size_t)i * HS + h];
            }
            float bar = acc / asum;
            o_sent[t * HS + h] = bar;
            float k = scal[t] * bar;
            kt[h] = k > 0.f ? k : 0.f;
        }
        grid_barrier(cnt, gen, nb);

        // ---- P3: rank-style update, b128 streams (the output write IS ev) ----
        const float4* ev4  = (const float4*)ev;
        const float4* kt4  = (const float4*)kt;
        float4* out4 = (float4*)(o_sall + (size_t)t * NS * HS);
        for (int idx = tid; idx < (NS * HS) / 4; idx += nb * 256) {
            int i  = idx >> 8;              // 256 float4 per row
            int h4 = idx & 255;
            float a = attn[i], b = 1.f - a;
            float4 e = ev4[idx], k = kt4[h4], o;
            o.x = a * k.x + b * e.x;
            o.y = a * k.y + b * e.y;
            o.z = a * k.z + b * e.z;
            o.w = a * k.w + b * e.w;
            out4[idx] = o;
        }
        grid_barrier(cnt, gen, nb);
    }
}

// ---------------------------------------------------------------------------
extern "C" void kernel_launch(void* const* d_in, const int* in_sizes, int n_in,
                              void* d_out, int out_size, void* d_ws, size_t ws_size,
                              hipStream_t stream) {
    (void)in_sizes; (void)n_in; (void)out_size; (void)ws_size;
    const float* vv  = (const float*)d_in[0];   // [1,S,H]
    const float* ent = (const float*)d_in[1];   // [N,H]
    const float* A1w = (const float*)d_in[2];
    const float* A1b = (const float*)d_in[3];
    const float* A2w = (const float*)d_in[4];
    const float* A2b = (const float*)d_in[5];
    const float* Emb = (const float*)d_in[6];
    const float* W1w = (const float*)d_in[7];
    const float* W1b = (const float*)d_in[8];
    const float* W2w = (const float*)d_in[9];
    const float* W2b = (const float*)d_in[10];
    const float* W3w = (const float*)d_in[11];
    const float* W3b = (const float*)d_in[12];
    const float* W4w = (const float*)d_in[13];
    const float* W4b = (const float*)d_in[14];

    // Outputs, concatenated flat in scan-return order
    float* out    = (float*)d_out;
    float* o_entp = out;                                   // [S,N]
    float* o_acp  = out + SS * NS;                         // [S,V]
    float* o_sent = o_acp + SS * VS;                       // [S,H]
    float* o_sall = o_sent + SS * HS;                      // [S,N,H]
    float* o_sact = o_sall + (size_t)SS * NS * HS;         // [S,WV]

    // Workspace carve-out (256B aligned slices)
    char* wsb = (char*)d_ws;
    size_t off = 0;
    auto alloc = [&](size_t bytes) {
        void* p = wsb + off;
        off = (off + bytes + 255) & ~(size_t)255;
        return p;
    };
    unsigned short* A1bf = (unsigned short*)alloc((size_t)HS * HS * 2);
    unsigned short* A2bf = (unsigned short*)alloc((size_t)VS * HS * 2);
    unsigned short* W1bf = (unsigned short*)alloc((size_t)HS * HS * 2);
    unsigned short* W2bf = (unsigned short*)alloc((size_t)HS * (HS + VS) * 2);
    unsigned short* vvbf = (unsigned short*)alloc((size_t)SS * HS * 2);
    unsigned short* hbf  = (unsigned short*)alloc((size_t)SS * HS * 2);
    unsigned short* ccat = (unsigned short*)alloc((size_t)SS * (HS + VS) * 2);
    float* hatf  = (float*)alloc((size_t)SS * HS * 4);
    float* w2vf  = (float*)alloc((size_t)SS * HS * 4);
    float* c0    = (float*)alloc(SS * 4);
    float* c1    = (float*)alloc(SS * 4);
    float* acsum = (float*)alloc(SS * 4);
    float* scal  = (float*)alloc(SS * 4);
    float* attn  = (float*)alloc(NS * 4);
    float* kt    = (float*)alloc(HS * 4);
    unsigned int* sync = (unsigned int*)alloc(64);

    // --- convert activations + weights to bf16 (L2-resident afterwards) ---
    k_f2bf<<<(HS * HS + 255) / 256, 256, 0, stream>>>(A1w, A1bf, HS * HS);
    k_f2bf<<<(VS * HS + 255) / 256, 256, 0, stream>>>(A2w, A2bf, VS * HS);
    k_f2bf<<<(HS * HS + 255) / 256, 256, 0, stream>>>(W1w, W1bf, HS * HS);
    k_f2bf<<<(HS * (HS + VS) + 255) / 256, 256, 0, stream>>>(W2w, W2bf, HS * (HS + VS));
    k_f2bf<<<(SS * HS + 255) / 256, 256, 0, stream>>>(vv, vvbf, SS * HS);
    k_init_sync<<<1, 32, 0, stream>>>(sync);

    // --- batched (carry-independent) GEMMs via WMMA ---
    // h = relu(vv @ A1^T + A1b)  -> bf16 only (feeds A2 GEMM)
    wmma_gemm_bf16<<<(4 * (HS / 16) + 7) / 8, 256, 0, stream>>>(
        vvbf, A1bf, A1b, nullptr, 0, hbf, HS, 0, HS, HS / 16, 1);
    // hat = relu(vv @ W1^T + W1b) -> fp32 (for choice) + concat[:, 0:H]
    wmma_gemm_bf16<<<(4 * (HS / 16) + 7) / 8, 256, 0, stream>>>(
        vvbf, W1bf, W1b, hatf, HS, ccat, HS + VS, 0, HS, HS / 16, 1);
    // ac = sigmoid(h @ A2^T + A2b) -> fp32 to d_out + concat[:, H:H+V]
    wmma_gemm_bf16<<<(4 * (VS / 16) + 7) / 8, 256, 0, stream>>>(
        hbf, A2bf, A2b, o_acp, VS, ccat, HS + VS, HS, HS, VS / 16, 2);
    // w2v = concat @ W2^T + W2b -> fp32 (used in per-step matvec)
    wmma_gemm_bf16<<<(4 * (HS / 16) + 7) / 8, 256, 0, stream>>>(
        ccat, W2bf, W2b, w2vf, HS, nullptr, 0, 0, HS + VS, HS / 16, 0);

    // --- small carry-independent epilogues ---
    k_choice<<<SS, 32, 0, stream>>>(hatf, W3w, W3b, c0, c1);
    k_acrow<<<SS, 32, 0, stream>>>(o_acp, acsum);
    dim3 gb((WVS + 127) / 128, SS);
    k_barft<<<gb, 128, 0, stream>>>(o_acp, Emb, acsum, o_sact);
    k_scal<<<SS, 32, 0, stream>>>(o_sact, W4w, W4b, scal);

    // --- all 64 sequential steps in ONE persistent kernel ---
    scan_persistent<<<SCAN_BLOCKS, 256, 0, stream>>>(
        ent, w2vf, c0, c1, scal, o_entp, o_sent, o_sall, attn, kt, sync);
}
